// PACT_84585085928013
// MI455X (gfx1250) — compile-verified, matrix operand order checked by reading, not run on hardware
//
#include <hip/hip_runtime.h>
#include <hip/hip_bf16.h>
#include <math.h>

#define BB 2
#define XX 384
#define YY 384
#define TT 4
#define CC 64
#define MAX_SHIFT 2.5f
#define GAMMA_ 0.8f
#define MAPN (BB * TT * YY * XX)   // 1,179,648 cells

typedef __attribute__((ext_vector_type(16))) _Float16 v16h;
typedef __attribute__((ext_vector_type(8)))  float    v8f;

// ---------------------------------------------------------------- init
// map[0..MAPN) = -1 ; sums[0..64] = 0 (64 center sums + 1 energy sum)
__global__ __launch_bounds__(256) void init_kernel(int* map, float* sums) {
    int idx = blockIdx.x * 256 + threadIdx.x;
    const int total = MAPN + 65;
    for (; idx < total; idx += gridDim.x * 256) {
        if (idx < MAPN) map[idx] = -1;
        else            sums[idx - MAPN] = 0.0f;
    }
}

// ---------------------------------------------------------------- build map
__global__ __launch_bounds__(256) void build_kernel(const int* __restrict__ coords,
                                                    int* __restrict__ map, int n) {
    int i = blockIdx.x * 256 + threadIdx.x;
    if (i >= n) return;
    int b = coords[4 * i + 0];
    int x = coords[4 * i + 1];
    int y = coords[4 * i + 2];
    int t = coords[4 * i + 3];
    map[((b * TT + t) * YY + y) * XX + x] = i;
}

// ---------------------------------------------------------------- column sums via WMMA
// center_sum[c] += sum_j feats[j][c], computed as ones(16x32) x B(32x16) per wmma.
// Lane L (L&15 = channel col within a 16-wide group) supplies 16 rows of that column.
__global__ __launch_bounds__(256) void center_kernel(const float* __restrict__ feats,
                                                     float* __restrict__ center_sum, int n) {
    const int lane  = threadIdx.x & 31;
    const int wid   = threadIdx.x >> 5;
    const int waves = gridDim.x * 8;
    const int gwave = blockIdx.x * 8 + wid;
    const int chunks = n >> 5;           // 32-row chunks
    const int half = lane >> 4;          // 0: rows k, 1: rows 16+k
    const int col  = lane & 15;

    v16h ones;
#pragma unroll
    for (int k = 0; k < 16; ++k) ones[k] = (_Float16)1.0f;

    v8f acc0 = {}, acc1 = {}, acc2 = {}, acc3 = {};

    for (int c = gwave; c < chunks; c += waves) {
        const float* p = feats + (size_t)(c * 32 + half * 16) * CC + col;
        v16h b0, b1, b2, b3;
#pragma unroll
        for (int k = 0; k < 16; ++k) {
            const float* q = p + (size_t)k * CC;
            b0[k] = (_Float16)q[0];
            b1[k] = (_Float16)q[16];
            b2[k] = (_Float16)q[32];
            b3[k] = (_Float16)q[48];
        }
        acc0 = __builtin_amdgcn_wmma_f32_16x16x32_f16(false, ones, false, b0, (short)0, acc0, false, false);
        acc1 = __builtin_amdgcn_wmma_f32_16x16x32_f16(false, ones, false, b1, (short)0, acc1, false, false);
        acc2 = __builtin_amdgcn_wmma_f32_16x16x32_f16(false, ones, false, b2, (short)0, acc2, false, false);
        acc3 = __builtin_amdgcn_wmma_f32_16x16x32_f16(false, ones, false, b3, (short)0, acc3, false, false);
    }

    __shared__ float smem[8 * 64];
    if (lane < 16) {
        smem[wid * 64 +  0 + col] = acc0[0];
        smem[wid * 64 + 16 + col] = acc1[0];
        smem[wid * 64 + 32 + col] = acc2[0];
        smem[wid * 64 + 48 + col] = acc3[0];
    }
    __syncthreads();
    int tid = threadIdx.x;
    if (tid < 64) {
        float s = 0.0f;
#pragma unroll
        for (int w = 0; w < 8; ++w) s += smem[w * 64 + tid];
        atomicAdd(&center_sum[tid], s);
    }
    // tail rows (n not multiple of 32) — scalar fallback
    int tail = chunks * 32;
    int row  = tail + blockIdx.x * 256 + threadIdx.x;
    if (row < n) {
        const float* fr = feats + (size_t)row * CC;
        for (int c = 0; c < CC; ++c) atomicAdd(&center_sum[c], fr[c]);
    }
}

// ---------------------------------------------------------------- gather: cons + energy
__global__ __launch_bounds__(256) void gather_kernel(const int* __restrict__ coords,
                                                     const float* __restrict__ feats,
                                                     const float* __restrict__ vel,
                                                     const int* __restrict__ map,
                                                     float* __restrict__ cons,
                                                     float* __restrict__ energy,
                                                     float* __restrict__ energy_sum, int n) {
    int j = blockIdx.x * 256 + threadIdx.x;
    float en = 0.0f;
    if (j < n) {
        int b  = coords[4 * j + 0];
        int xj = coords[4 * j + 1];
        int yj = coords[4 * j + 2];
        int tj = coords[4 * j + 3];

        float4 acc[16];
#pragma unroll
        for (int q = 0; q < 16; ++q) acc[q] = make_float4(0.f, 0.f, 0.f, 0.f);
        float wsum = 0.0f;
        float xjf = (float)xj, yjf = (float)yj;

        if (tj > 0) {
            const int* mslice = map + (size_t)((b * TT + (tj - 1)) * YY) * XX;
            for (int dy = -3; dy <= 3; ++dy) {
                int ny = yj + dy;
                if ((unsigned)ny >= (unsigned)YY) continue;
                const int* mrow = mslice + ny * XX;
                __builtin_prefetch(mrow + (xj < 3 ? 0 : xj - 3), 0, 0);
                for (int dx = -3; dx <= 3; ++dx) {
                    int nx = xj + dx;
                    if ((unsigned)nx >= (unsigned)XX) continue;
                    int i = mrow[nx];
                    if (i < 0) continue;
                    float vx = vel[2 * i + 0];
                    float vy = vel[2 * i + 1];
                    vx = fminf(fmaxf(vx, -MAX_SHIFT), MAX_SHIFT);
                    vy = fminf(fmaxf(vy, -MAX_SHIFT), MAX_SHIFT);
                    float xt = (float)nx + vx;
                    float yt = (float)ny + vy;
                    float x0 = floorf(xt), y0 = floorf(yt);
                    bool hx = (xjf == x0) || (xjf == x0 + 1.0f);
                    bool hy = (yjf == y0) || (yjf == y0 + 1.0f);
                    if (hx && hy) {
                        float ddx = xt - xjf, ddy = yt - yjf;
                        float w = expf(-(ddx * ddx + ddy * ddy));
                        wsum += w;
                        const float4* fi = (const float4*)(feats + (size_t)i * CC);
#pragma unroll
                        for (int q = 0; q < 16; ++q) {
                            float4 v = fi[q];
                            acc[q].x += w * v.x; acc[q].y += w * v.y;
                            acc[q].z += w * v.z; acc[q].w += w * v.w;
                        }
                    }
                }
            }
        }
        float inv = 1.0f / fmaxf(wsum, 1e-6f);   // GAIN = 1.0
        float sad = 0.0f, sab = 0.0f;
        const float4* fj = (const float4*)(feats + (size_t)j * CC);
#pragma unroll
        for (int q = 0; q < 16; ++q) {
            float4 f = fj[q];
            float ax = acc[q].x * inv, ay = acc[q].y * inv;
            float az = acc[q].z * inv, aw = acc[q].w * inv;
            sad += fabsf(ax - f.x) + fabsf(ay - f.y) + fabsf(az - f.z) + fabsf(aw - f.w);
            sab += fabsf(f.x) + fabsf(f.y) + fabsf(f.z) + fabsf(f.w);
            en  += f.x * f.x + f.y * f.y + f.z * f.z + f.w * f.w;
        }
        float diff = sad / fmaxf(sab, 1e-6f);
        cons[j]   = expf(-diff);
        energy[j] = en;
    }
    // block reduce energy for the global mean
    __shared__ float sred[256];
    sred[threadIdx.x] = en;
    __syncthreads();
    for (int s = 128; s > 0; s >>= 1) {
        if (threadIdx.x < s) sred[threadIdx.x] += sred[threadIdx.x + s];
        __syncthreads();
    }
    if (threadIdx.x == 0) atomicAdd(energy_sum, sred[0]);
}

// ---------------------------------------------------------------- finalize
__global__ __launch_bounds__(256) void final_kernel(const float* __restrict__ feats,
                                                    const float* __restrict__ cons,
                                                    const float* __restrict__ energy,
                                                    const float* __restrict__ center_sum,
                                                    const float* __restrict__ energy_sum,
                                                    float* __restrict__ out_feats,
                                                    float* __restrict__ out_score, int n) {
    int j = blockIdx.x * 256 + threadIdx.x;
    if (j >= n) return;
    float invn = 1.0f / (float)n;
    float me = energy_sum[0] * invn;
    float e  = energy[j] / (me + 1e-6f);
    e = fminf(fmaxf(e, 0.0f), 4.0f) * 0.25f;
    float t = cons[j] * e;
    out_score[j] = t;
    float g = GAMMA_ * t;
    const float4* fj  = (const float4*)(feats + (size_t)j * CC);
    float4*       oj  = (float4*)(out_feats + (size_t)j * CC);
    const float4* cen = (const float4*)center_sum;
#pragma unroll
    for (int q = 0; q < 16; ++q) {
        float4 f = fj[q];
        float4 c = cen[q];
        float4 o;
        o.x = f.x + g * (f.x - c.x * invn);
        o.y = f.y + g * (f.y - c.y * invn);
        o.z = f.z + g * (f.z - c.z * invn);
        o.w = f.w + g * (f.w - c.w * invn);
        oj[q] = o;
    }
}

// ---------------------------------------------------------------- launch
extern "C" void kernel_launch(void* const* d_in, const int* in_sizes, int n_in,
                              void* d_out, int out_size, void* d_ws, size_t ws_size,
                              hipStream_t stream) {
    const int*   coords = (const int*)d_in[0];
    const float* feats  = (const float*)d_in[1];
    const float* vel    = (const float*)d_in[2];
    float* out = (float*)d_out;
    const int n = in_sizes[0] / 4;

    char* ws = (char*)d_ws;
    int*   map        = (int*)ws;                                   // MAPN ints
    float* cons       = (float*)(ws + (size_t)MAPN * 4);            // n floats
    float* energy     = cons + n;                                   // n floats
    float* sums       = energy + n;                                 // 65 floats
    float* center_sum = sums;                                       // [0..63]
    float* energy_sum = sums + 64;                                  // [64]

    float* out_feats = out;
    float* out_score = out + (size_t)n * CC;

    int blocksN   = (n + 255) / 256;
    int blocksIni = (MAPN + 65 + 255) / 256;

    init_kernel  <<<blocksIni, 256, 0, stream>>>(map, sums);
    build_kernel <<<blocksN,   256, 0, stream>>>(coords, map, n);
    center_kernel<<<512,       256, 0, stream>>>(feats, center_sum, n);
    gather_kernel<<<blocksN,   256, 0, stream>>>(coords, feats, vel, map,
                                                 cons, energy, energy_sum, n);
    final_kernel <<<blocksN,   256, 0, stream>>>(feats, cons, energy, center_sum,
                                                 energy_sum, out_feats, out_score, n);
}
